// MultiHeadAttention_29291676959197
// MI455X (gfx1250) — matrix operand order repaired and checked by reading, not compiled, not run on hardware
//
#include <hip/hip_runtime.h>
#include <hip/hip_bf16.h>

// MHA forward for MI455X (gfx1250, wave32, WMMA, async global->LDS with
// double-buffered pipelining).
// Pipeline: fp32->bf16 pre-convert (x, Wq, Wk, Wv, Wp) -> 3x bf16 GEMM
// (Q/K/V projections) -> flash-attention (bf16 WMMA, fp32 online softmax,
// bf16 out) -> bf16 GEMM (output projection, fp32 d_out).

#define N_EMBD 1024
#define N_HEADS 16
#define HEAD_D 64
#define BSZ 4
#define SEQ 2048

typedef __attribute__((ext_vector_type(16))) __bf16 v16bf;
typedef __attribute__((ext_vector_type(4)))  __bf16 v4bf;
typedef __attribute__((ext_vector_type(8)))  float  v8f;
typedef int v4i __attribute__((vector_size(16)));

union Frag { v16bf v; uint4 q[2]; };

__device__ __forceinline__ v8f wmma_bf16(const Frag& a, const Frag& b, v8f c) {
  return __builtin_amdgcn_wmma_f32_16x16x32_bf16(
      /*neg_a=*/false, a.v, /*neg_b=*/false, b.v,
      /*c_mod=*/(short)0, c, /*reuse_a=*/false, /*reuse_b=*/false);
}

// --- CDNA5 async global->LDS (ASYNCcnt) with safe fallback ------------------
#define HAS_ASYNC_LDS __has_builtin(__builtin_amdgcn_global_load_async_to_lds_b128)

__device__ __forceinline__ void copy16_g2l(const __bf16* g, __bf16* l) {
#if HAS_ASYNC_LDS
  __builtin_amdgcn_global_load_async_to_lds_b128(
      (__attribute__((address_space(1))) v4i*)(g),
      (__attribute__((address_space(3))) v4i*)(l), 0, 0);
#else
  *(uint4*)l = *(const uint4*)g;
#endif
}

// Wait until at most N async transfers remain outstanding (in-order
// completion per ISA: the oldest ones are the finished ones).
template <int N>
__device__ __forceinline__ void async_join() {
#if HAS_ASYNC_LDS
#if __has_builtin(__builtin_amdgcn_s_wait_asynccnt)
  __builtin_amdgcn_s_wait_asynccnt(N);
#else
  asm volatile("s_wait_asynccnt %0" ::"i"(N) : "memory");
#endif
#endif
}

// ---------------------------------------------------------------------------
// fp32 -> bf16 elementwise pre-convert (vectorized float4 -> 4x bf16).
// ---------------------------------------------------------------------------
__global__ __launch_bounds__(256) void f32_to_bf16(
    const float* __restrict__ in, __bf16* __restrict__ out, int n4) {
  const int i = blockIdx.x * blockDim.x + threadIdx.x;
  if (i < n4) {
    const float4 v = ((const float4*)in)[i];
    v4bf o; o[0] = (__bf16)v.x; o[1] = (__bf16)v.y;
            o[2] = (__bf16)v.z; o[3] = (__bf16)v.w;
    ((v4bf*)out)[i] = o;
  }
}

// ---------------------------------------------------------------------------
// GEMM: Y[M,N] = X[M,K](bf16) @ W[N,K](bf16)^T + bias[N](f32).
// 128 threads = 4 waves in a 2x2 grid; each wave owns a 32x64 patch
// (2 A frags + 4 B frags -> 8 WMMA per K-step). BM=64, BN=128, BK=32.
// LDS tiles are double-buffered: async loads for tile i+1 are issued before
// waiting on tile i (s_wait_asynccnt 6 keeps the younger batch in flight).
// All fragments are loaded into distinct registers before the WMMA chain so
// the scheduler can overlap ds_load latency with WMMA issue.
// ---------------------------------------------------------------------------
constexpr int BM = 64, BN = 128, BK = 32;
constexpr int LDT = BK + 8;       // 40 elems -> 80B row stride (16B multiple)
constexpr int STAGE_OPS = 6;      // async issues per thread per stage

template <typename OutT>
__global__ __launch_bounds__(128) void gemm_xwt_bias(
    const __bf16* __restrict__ X, const __bf16* __restrict__ W,
    const float* __restrict__ bias, OutT* __restrict__ Y,
    int M, int N, int K) {
  __shared__ alignas(16) __bf16 Xs[2][BM][LDT];   // 10 KB
  __shared__ alignas(16) __bf16 Ws[2][BN][LDT];   // 20 KB
  const int tid = threadIdx.x;
  const int lane = tid & 31, wave = tid >> 5;
  const int wr = wave >> 1, wc = wave & 1;        // 2 x 2 waves
  const int l16 = lane & 15, half = lane >> 4;
  const int m0 = blockIdx.y * BM, n0 = blockIdx.x * BN;

  // Stage one K-step (X: 64x32, W: 128x32 bf16) into buffer s.
  auto stage = [&](int s, int kk) {
#pragma unroll
    for (int i = 0; i < 2; ++i) {            // X: 256 b128 chunks / 128 thr
      const int c = tid + i * 128;
      copy16_g2l(X + (size_t)(m0 + (c >> 2)) * K + kk + (c & 3) * 8,
                 &Xs[s][c >> 2][(c & 3) * 8]);
    }
#pragma unroll
    for (int i = 0; i < 4; ++i) {            // W: 512 b128 chunks / 128 thr
      const int c = tid + i * 128;
      copy16_g2l(W + (size_t)(n0 + (c >> 2)) * K + kk + (c & 3) * 8,
                 &Ws[s][c >> 2][(c & 3) * 8]);
    }
  };

  v8f acc[2][4] = {};
  const int nIter = K / BK;
  stage(0, 0);
  for (int it = 0; it < nIter; ++it) {
    const int buf = it & 1;
    if (it + 1 < nIter) {
      stage(buf ^ 1, (it + 1) * BK);  // overlap next tile with this compute
      async_join<STAGE_OPS>();        // oldest 6 (this tile) are complete
    } else {
      async_join<0>();
    }
    __syncthreads();

    // A frags: lane m = l16; K elems {h8..h8+7, h8+16..h8+23}, h8 = half*8.
    Frag a0, a1;
    {
      const __bf16* p = &Xs[buf][wr * 32 + l16][half * 8];
      a0.q[0] = *(const uint4*)p;  a0.q[1] = *(const uint4*)(p + 16);
      const __bf16* q = &Xs[buf][wr * 32 + 16 + l16][half * 8];
      a1.q[0] = *(const uint4*)q;  a1.q[1] = *(const uint4*)(q + 16);
    }
    // B frags (distinct regs): lane n = l16; K elems {h8..h8+7, h8+16..h8+23}.
    Frag bfr[4];
#pragma unroll
    for (int db = 0; db < 4; ++db) {
      const __bf16* p = &Ws[buf][wc * 64 + db * 16 + l16][half * 8];
      bfr[db].q[0] = ((const uint4*)p)[0];
      bfr[db].q[1] = ((const uint4*)p)[2];
    }
#pragma unroll
    for (int db = 0; db < 4; ++db) {
      acc[0][db] = wmma_bf16(a0, bfr[db], acc[0][db]);
      acc[1][db] = wmma_bf16(a1, bfr[db], acc[1][db]);
    }
    __syncthreads();  // all waves done reading buf -> safe to refill it
  }

  // C layout: vgpr e -> row (half*8 + e), col l16.
  const int nb0 = n0 + wc * 64 + l16;
  const int mb0 = m0 + wr * 32 + half * 8;
#pragma unroll
  for (int s = 0; s < 2; ++s)
#pragma unroll
    for (int db = 0; db < 4; ++db) {
      const int n = nb0 + db * 16;
      const float bv = bias[n];
#pragma unroll
      for (int e = 0; e < 8; ++e)
        Y[(size_t)(mb0 + s * 16 + e) * N + n] = (OutT)(acc[s][db][e] + bv);
    }
}

// ---------------------------------------------------------------------------
// Flash attention: one (b,h,64-query tile) per block, 4 waves x 16 rows.
// K tiles staged async to LDS; V staged transposed [d][key]; P bounced
// through per-wave LDS (C layout -> A layout). bf16 output.
// ---------------------------------------------------------------------------
constexpr int QT = 64, KTILE = 64, PADK = HEAD_D + 8;  // 72 elems = 144B rows

__global__ __launch_bounds__(128) void attn_fwd(
    const __bf16* __restrict__ Q, const __bf16* __restrict__ Kg,
    const __bf16* __restrict__ Vg, __bf16* __restrict__ O) {
  __shared__ alignas(16) __bf16 Ks[KTILE][PADK];
  __shared__ alignas(16) __bf16 Vt[HEAD_D][PADK];
  __shared__ alignas(16) __bf16 Ps[4][16][PADK];

  const int tid = threadIdx.x, lane = tid & 31, wave = tid >> 5;
  const int l16 = lane & 15, half = lane >> 4;
  const int bh = blockIdx.y;
  const int b = bh >> 4, h = bh & (N_HEADS - 1);
  const int t0 = blockIdx.x * QT;

  const __bf16* Qb = Q  + (size_t)b * SEQ * N_EMBD + h * HEAD_D;
  const __bf16* Kb = Kg + (size_t)b * SEQ * N_EMBD + h * HEAD_D;
  const __bf16* Vb = Vg + (size_t)b * SEQ * N_EMBD + h * HEAD_D;

  Frag qa[2];
  {
    const __bf16* qrow = Qb + (size_t)(t0 + wave * 16 + l16) * N_EMBD + half * 8;
#pragma unroll
    for (int f = 0; f < 2; ++f) {
      qa[f].q[0] = *(const uint4*)(qrow + f * 32);
      qa[f].q[1] = *(const uint4*)(qrow + f * 32 + 16);
    }
  }

  float mrow[8], lrow[8];
  v8f accO[4] = {};
#pragma unroll
  for (int e = 0; e < 8; ++e) { mrow[e] = -3.0e38f; lrow[e] = 0.f; }

  const int krow0 = tid >> 1, kseg = (tid & 1) * 32;
  const int jmax = t0 >> 6;
  for (int j = 0; j <= jmax; ++j) {
    __syncthreads();
    {  // K tile async (2 b128/thread); V tile transposed synchronously.
#pragma unroll
      for (int i = 0; i < 2; ++i)
        copy16_g2l(Kb + (size_t)(j * KTILE + krow0) * N_EMBD + kseg + i * 8,
                   &Ks[krow0][kseg + i * 8]);
      const __bf16* vrow = Vb + (size_t)(j * KTILE + krow0) * N_EMBD + kseg;
      union { uint4 q; __bf16 e[8]; } vv;
#pragma unroll
      for (int i = 0; i < 4; ++i) {
        vv.q = ((const uint4*)vrow)[i];
#pragma unroll
        for (int d = 0; d < 8; ++d) Vt[kseg + i * 8 + d][krow0] = vv.e[d];
      }
      if (j < jmax) {  // prefetch next tile (global_prefetch_b8)
        __builtin_prefetch(
            Kb + (size_t)((j + 1) * KTILE + krow0) * N_EMBD + kseg, 0, 0);
        __builtin_prefetch(
            Vb + (size_t)((j + 1) * KTILE + krow0) * N_EMBD + kseg, 0, 0);
      }
    }
    async_join<0>();
    __syncthreads();

    // S = Q K^T : load all 8 B-frags into distinct regs, then WMMA chain.
    Frag kf[4][2];
#pragma unroll
    for (int nb = 0; nb < 4; ++nb)
#pragma unroll
      for (int kc = 0; kc < 2; ++kc) {
        const __bf16* bp = &Ks[nb * 16 + l16][kc * 32 + half * 8];
        kf[nb][kc].q[0] = ((const uint4*)bp)[0];
        kf[nb][kc].q[1] = ((const uint4*)bp)[2];
      }
    v8f s[4];
#pragma unroll
    for (int nb = 0; nb < 4; ++nb) {
      v8f sa = {};
      sa = wmma_bf16(qa[0], kf[nb][0], sa);
      sa = wmma_bf16(qa[1], kf[nb][1], sa);
      s[nb] = sa;
    }

    // scale + causal mask + online softmax (rows = half*8 + e).
    const int qi0 = t0 + wave * 16 + half * 8;
#pragma unroll
    for (int e = 0; e < 8; ++e) {
#pragma unroll
      for (int nb = 0; nb < 4; ++nb) {
        float v = s[nb][e] * 0.125f;  // 1/sqrt(64)
        if (j * KTILE + nb * 16 + l16 > qi0 + e) v = -3.0e38f;
        s[nb][e] = v;
      }
      float mx = fmaxf(fmaxf(s[0][e], s[1][e]), fmaxf(s[2][e], s[3][e]));
#pragma unroll
      for (int off = 1; off < 16; off <<= 1)
        mx = fmaxf(mx, __shfl_xor(mx, off, 32));
      const float mnew = fmaxf(mrow[e], mx);
      const float corr = __expf(mrow[e] - mnew);
      mrow[e] = mnew;
      float ps = 0.f;
#pragma unroll
      for (int nb = 0; nb < 4; ++nb) {
        const float p = __expf(s[nb][e] - mnew);
        s[nb][e] = p;
        ps += p;
      }
#pragma unroll
      for (int off = 1; off < 16; off <<= 1)
        ps += __shfl_xor(ps, off, 32);
      lrow[e] = lrow[e] * corr + ps;
#pragma unroll
      for (int db = 0; db < 4; ++db) accO[db][e] *= corr;
    }

    // P: C layout -> A layout via per-wave LDS (same-wave DS ops in order).
#pragma unroll
    for (int nb = 0; nb < 4; ++nb)
#pragma unroll
      for (int e = 0; e < 8; ++e)
        Ps[wave][half * 8 + e][nb * 16 + l16] = (__bf16)s[nb][e];

    Frag pa[2];
#pragma unroll
    for (int kc = 0; kc < 2; ++kc) {
      const __bf16* pp = &Ps[wave][l16][kc * 32 + half * 8];
      pa[kc].q[0] = *(const uint4*)pp;
      pa[kc].q[1] = *(const uint4*)(pp + 16);
    }
    // O += P V : load all 8 V-frags into distinct regs, then WMMA chain.
    Frag vf[4][2];
#pragma unroll
    for (int db = 0; db < 4; ++db)
#pragma unroll
      for (int kc = 0; kc < 2; ++kc) {
        const __bf16* vp = &Vt[db * 16 + l16][kc * 32 + half * 8];
        vf[db][kc].q[0] = ((const uint4*)vp)[0];
        vf[db][kc].q[1] = ((const uint4*)vp)[2];
      }
#pragma unroll
    for (int db = 0; db < 4; ++db) {
      accO[db] = wmma_bf16(pa[0], vf[db][0], accO[db]);
      accO[db] = wmma_bf16(pa[1], vf[db][1], accO[db]);
    }
  }

  // Write attention output as bf16 [B, T, h*64 + d].
#pragma unroll
  for (int db = 0; db < 4; ++db) {
    const int d = db * 16 + l16;
#pragma unroll
    for (int e = 0; e < 8; ++e) {
      const int t = t0 + wave * 16 + half * 8 + e;
      O[((size_t)b * SEQ + t) * N_EMBD + h * HEAD_D + d] =
          (__bf16)(accO[db][e] / lrow[e]);
    }
  }
}

// ---------------------------------------------------------------------------
extern "C" void kernel_launch(void* const* d_in, const int* in_sizes, int n_in,
                              void* d_out, int out_size, void* d_ws, size_t ws_size,
                              hipStream_t stream) {
  const float* x  = (const float*)d_in[0];
  const float* Wk = (const float*)d_in[1];
  const float* bk = (const float*)d_in[2];
  const float* Wq = (const float*)d_in[3];
  const float* bq = (const float*)d_in[4];
  const float* Wv = (const float*)d_in[5];
  const float* bv = (const float*)d_in[6];
  const float* Wp = (const float*)d_in[7];
  const float* bp = (const float*)d_in[8];

  const int M = BSZ * SEQ;                 // 8192
  const size_t xN  = (size_t)M * N_EMBD;   // 8M elems
  const size_t wN  = (size_t)N_EMBD * N_EMBD;  // 1M elems

  char* ws = (char*)d_ws;
  __bf16* xb  = (__bf16*)(ws);                          // 16 MB
  __bf16* wqb = (__bf16*)(ws + 2 * xN);                 // 2 MB each
  __bf16* wkb = wqb + wN;
  __bf16* wvb = wkb + wN;
  __bf16* wpb = wvb + wN;
  __bf16* qb  = wpb + wN;                               // 16 MB each
  __bf16* kb  = qb + xN;
  __bf16* vb  = kb + xN;
  __bf16* attnb = vb + xN;                              // 16 MB

  // One-shot fp32 -> bf16 conversion (x + 4 weight matrices).
  f32_to_bf16<<<(int)(xN / 4 + 255) / 256, 256, 0, stream>>>(x,  xb,  (int)(xN / 4));
  f32_to_bf16<<<(int)(wN / 4 + 255) / 256, 256, 0, stream>>>(Wq, wqb, (int)(wN / 4));
  f32_to_bf16<<<(int)(wN / 4 + 255) / 256, 256, 0, stream>>>(Wk, wkb, (int)(wN / 4));
  f32_to_bf16<<<(int)(wN / 4 + 255) / 256, 256, 0, stream>>>(Wv, wvb, (int)(wN / 4));
  f32_to_bf16<<<(int)(wN / 4 + 255) / 256, 256, 0, stream>>>(Wp, wpb, (int)(wN / 4));

  dim3 gg(N_EMBD / BN, M / BM);  // (8, 128)
  gemm_xwt_bias<__bf16><<<gg, 128, 0, stream>>>(xb, wqb, bq, qb, M, N_EMBD, N_EMBD);
  gemm_xwt_bias<__bf16><<<gg, 128, 0, stream>>>(xb, wkb, bk, kb, M, N_EMBD, N_EMBD);
  gemm_xwt_bias<__bf16><<<gg, 128, 0, stream>>>(xb, wvb, bv, vb, M, N_EMBD, N_EMBD);

  attn_fwd<<<dim3(SEQ / QT, BSZ * N_HEADS), 128, 0, stream>>>(qb, kb, vb, attnb);

  gemm_xwt_bias<float><<<gg, 128, 0, stream>>>(attnb, wpb, bp, (float*)d_out,
                                               M, N_EMBD, N_EMBD);
  (void)in_sizes; (void)n_in; (void)out_size; (void)ws_size;
}
